// Custom_RNN_6768868459204
// MI455X (gfx1250) — matrix operand order repaired and checked by reading, not compiled, not run on hardware
//
#include <hip/hip_runtime.h>
#include <math.h>

#define VV 50257
#define EE 1024
#define HH 1024
#define BB 2048

typedef __attribute__((ext_vector_type(16))) __bf16 v16bf;
typedef __attribute__((ext_vector_type(8)))  __bf16 bf8;
typedef __attribute__((ext_vector_type(4)))  __bf16 bf4;
typedef __attribute__((ext_vector_type(8)))  float  v8f;

__device__ __forceinline__ __bf16 f2bf(float f) {
    union { float f; unsigned u; } v; v.f = f;
    unsigned r = v.u + 0x7FFFu + ((v.u >> 16) & 1u);   // round-to-nearest-even
    unsigned short h = (unsigned short)(r >> 16);
    return __builtin_bit_cast(__bf16, h);
}

union Frag { v16bf v; bf8 h[2]; };

// Split workgroup barrier + LDS fence (mnemonics verified in gfx1250 disasm).
#define LDS_FENCE()   asm volatile("s_wait_dscnt 0x0" ::: "memory")
#define BAR_SIGNAL()  asm volatile("s_barrier_signal -1" ::: "memory")
#define BAR_WAIT()    asm volatile("s_barrier_wait -1" ::: "memory")

// ---------------------------------------------------------------------------
// Embedding gather: xe[b,:] = emb[x[b],:]
// ---------------------------------------------------------------------------
__global__ void gather_rows(const int* __restrict__ x,
                            const float* __restrict__ emb,
                            float* __restrict__ xe) {
    int b = blockIdx.x;
    const float4* src = (const float4*)(emb + (size_t)x[b] * EE);
    float4* dst = (float4*)(xe + (size_t)b * EE);
    dst[threadIdx.x] = src[threadIdx.x];          // 256 threads * 4 floats = 1024
}

// c_new = tanh(pre) * (c + tanh(cand));  hc = h + c_new
__global__ void cell_ew(const float* __restrict__ pre, const float* __restrict__ cand,
                        const float* __restrict__ c,   const float* __restrict__ h,
                        float* __restrict__ c_new, float* __restrict__ hc, int n) {
    int i = blockIdx.x * blockDim.x + threadIdx.x;
    if (i < n) {
        float t  = tanhf(pre[i]);
        float cn = t * (c[i] + tanhf(cand[i]));
        c_new[i] = cn;
        hc[i]    = h[i] + cn;
    }
}

// h_new = tanh(opre) * tanh(c_new)
__global__ void hnew_ew(const float* __restrict__ opre, const float* __restrict__ c_new,
                        float* __restrict__ h_new, int n) {
    int i = blockIdx.x * blockDim.x + threadIdx.x;
    if (i < n) h_new[i] = tanhf(opre[i]) * tanhf(c_new[i]);
}

// ---------------------------------------------------------------------------
// Staging split: global-load phase (batched b128 into regs) vs LDS-store phase,
// so the loads can be issued between s_barrier_signal and s_barrier_wait.
// ---------------------------------------------------------------------------
template <bool FULL>
__device__ __forceinline__ void stage_load(const float* __restrict__ A,
                                           const float* __restrict__ W,
                                           int N, int K, int blockM, int blockN,
                                           int kk, int tid,
                                           float4 fa[4], float4 fw[4]) {
    const int c4 = (tid & 7) * 4;
    const int r0 = tid >> 3;
#pragma unroll
    for (int i = 0; i < 4; ++i) {
        int r = r0 + i * 32;
        fa[i] = *(const float4*)(A + (size_t)(blockM + r) * K + kk + c4);
    }
#pragma unroll
    for (int i = 0; i < 4; ++i) {
        int r  = r0 + i * 32;
        int gn = blockN + r;
        if (FULL || gn < N) fw[i] = *(const float4*)(W + (size_t)gn * K + kk + c4);
        else                fw[i] = make_float4(0.f, 0.f, 0.f, 0.f);
    }
}

__device__ __forceinline__ void stage_store(const float4 fa[4], const float4 fw[4],
                                            int tid,
                                            __bf16* __restrict__ As,
                                            __bf16* __restrict__ Ws) {
    const int c4 = (tid & 7) * 4;
    const int r0 = tid >> 3;
#pragma unroll
    for (int i = 0; i < 4; ++i) {
        int r = r0 + i * 32;
        bf4 pa, pw;
        pa.x = f2bf(fa[i].x); pa.y = f2bf(fa[i].y); pa.z = f2bf(fa[i].z); pa.w = f2bf(fa[i].w);
        pw.x = f2bf(fw[i].x); pw.y = f2bf(fw[i].y); pw.z = f2bf(fw[i].z); pw.w = f2bf(fw[i].w);
        *(bf4*)&As[r * 40 + c4] = pa;
        *(bf4*)&Ws[r * 40 + c4] = pw;
    }
}

// Consume one staged K=32 slice: 2x4 tiles of v_wmma_f32_16x16x32_bf16 per wave.
__device__ __forceinline__ void compute32(const __bf16* __restrict__ As,
                                          const __bf16* __restrict__ Ws,
                                          v8f acc[2][4],
                                          int wave_m, int wave_n, int l16, int half) {
    Frag a[2];
#pragma unroll
    for (int mt = 0; mt < 2; ++mt) {
        int row = wave_m * 32 + mt * 16 + l16;
        a[mt].h[0] = *(const bf8*)&As[row * 40 + 8 * half];       // K 8h..8h+7
        a[mt].h[1] = *(const bf8*)&As[row * 40 + 16 + 8 * half];  // K 16+8h..
    }
#pragma unroll
    for (int nt = 0; nt < 4; ++nt) {
        Frag b;
        int col = wave_n * 64 + nt * 16 + l16;
        b.h[0] = *(const bf8*)&Ws[col * 40 + 8 * half];           // K 8h..8h+7
        b.h[1] = *(const bf8*)&Ws[col * 40 + 16 + 8 * half];      // K 16+8h..
#pragma unroll
        for (int mt = 0; mt < 2; ++mt)
            acc[mt][nt] = __builtin_amdgcn_wmma_f32_16x16x32_bf16(
                false, a[mt].v, false, b.v, (short)0, acc[mt][nt], false, false);
    }
}

// One (A, W^T) pair with LDS double buffering; global loads of slice s+1 are
// issued between barrier signal and wait so load latency hides behind the
// barrier and the WMMAs of slice s.
template <bool FULL>
__device__ __forceinline__ void gemm_pair(const float* __restrict__ A,
                                          const float* __restrict__ W,
                                          int N, int K, int blockM, int blockN, int tid,
                                          __bf16* __restrict__ As0, __bf16* __restrict__ As1,
                                          __bf16* __restrict__ Ws0, __bf16* __restrict__ Ws1,
                                          v8f acc[2][4],
                                          int wave_m, int wave_n, int l16, int half) {
    float4 fa[4], fw[4];
    stage_load<FULL>(A, W, N, K, blockM, blockN, 0, tid, fa, fw);
    stage_store(fa, fw, tid, As0, Ws0);
    const int nk = K >> 5;
    for (int s = 0; s < nk; ++s) {
        LDS_FENCE();                       // our ds ops visible before signaling
        BAR_SIGNAL();
        const bool more = (s + 1 < nk);
        if (more)                          // loads in flight across the barrier
            stage_load<FULL>(A, W, N, K, blockM, blockN, (s + 1) << 5, tid, fa, fw);
        BAR_WAIT();
        if (more)
            stage_store(fa, fw, tid, (s & 1) ? As0 : As1, (s & 1) ? Ws0 : Ws1);
        compute32((s & 1) ? As1 : As0, (s & 1) ? Ws1 : Ws0, acc, wave_m, wave_n, l16, half);
    }
    LDS_FENCE();                           // protect buffers for the next pair
    BAR_SIGNAL();
    BAR_WAIT();
}

// Whole-block GEMM body; FULL=true compiles out every N-guard.
template <bool FULL>
__device__ __forceinline__ void gemm_body(const float* A0, const float* W0,
                                          const float* A1, const float* W1,
                                          const float* A2, const float* W2,
                                          int npairs,
                                          const float* bias1, const float* bias2,
                                          float* C, int N, int K, int ntC,
                                          int blockM, int blockN, int tid,
                                          __bf16* As0, __bf16* As1,
                                          __bf16* Ws0, __bf16* Ws1) {
    const int lane   = tid & 31;
    const int wid    = tid >> 5;
    const int wave_m = wid & 3;            // 4 waves over M (4*32 = 128)
    const int wave_n = wid >> 2;           // 2 waves over N (2*64 = 128)
    const int l16    = lane & 15;
    const int half   = lane >> 4;

    v8f acc[2][4];
#pragma unroll
    for (int mt = 0; mt < 2; ++mt)
#pragma unroll
        for (int nt = 0; nt < 4; ++nt)
            acc[mt][nt] = v8f{0.f, 0.f, 0.f, 0.f, 0.f, 0.f, 0.f, 0.f};

    gemm_pair<FULL>(A0, W0, N, K, blockM, blockN, tid, As0, As1, Ws0, Ws1,
                    acc, wave_m, wave_n, l16, half);
    if (npairs > 1)
        gemm_pair<FULL>(A1, W1, N, K, blockM, blockN, tid, As0, As1, Ws0, Ws1,
                        acc, wave_m, wave_n, l16, half);
    if (npairs > 2)
        gemm_pair<FULL>(A2, W2, N, K, blockM, blockN, tid, As0, As1, Ws0, Ws1,
                        acc, wave_m, wave_n, l16, half);

    // ---- epilogue: bias + store (guards compiled out when FULL) ----
    float bcol[4];
#pragma unroll
    for (int nt = 0; nt < 4; ++nt) {
        int gcol = blockN + wave_n * 64 + nt * 16 + l16;
        float bv = 0.f;
        if (FULL || gcol < N) {
            if (bias1) bv += bias1[gcol];
            if (bias2) bv += bias2[gcol];
        }
        bcol[nt] = bv;
    }
#pragma unroll
    for (int mt = 0; mt < 2; ++mt) {
#pragma unroll
        for (int nt = 0; nt < 4; ++nt) {
            int gcol = blockN + wave_n * 64 + nt * 16 + l16;
            if (FULL || gcol < N) {
                int rbase = blockM + wave_m * 32 + mt * 16 + half * 8;
                if (ntC) {                 // streaming output: keep L2 for Wy/A
#pragma unroll
                    for (int i = 0; i < 8; ++i)
                        __builtin_nontemporal_store(acc[mt][nt][i] + bcol[nt],
                                                    &C[(size_t)(rbase + i) * N + gcol]);
                } else {
#pragma unroll
                    for (int i = 0; i < 8; ++i)
                        C[(size_t)(rbase + i) * N + gcol] = acc[mt][nt][i] + bcol[nt];
                }
            }
        }
    }
}

// ---------------------------------------------------------------------------
// Generic GEMM: C[2048, N] = sum_p A_p[2048,K] * W_p[N,K]^T + bias1 + bias2
// Block tile 128x128, 8 waves (wave32), wave tile 32x64, K-step 32.
// Grid mapping: blockIdx.x -> M (16 blocks, fast), blockIdx.y -> N.
// x-fastest launch makes the 16 M-blocks that share one 512 KB W tile adjacent,
// so W streams from HBM once and A (8 MB) stays resident in the 192 MB L2.
// ---------------------------------------------------------------------------
__global__ __launch_bounds__(256)
void gemm_bf16(const float* __restrict__ A0, const float* __restrict__ W0,
               const float* __restrict__ A1, const float* __restrict__ W1,
               const float* __restrict__ A2, const float* __restrict__ W2,
               int npairs,
               const float* __restrict__ bias1, const float* __restrict__ bias2,
               float* __restrict__ C, int N, int K, int ntC) {
    __shared__ __align__(16) __bf16 As[2][128 * 40];   // stride 40 bf16 = 80B rows
    __shared__ __align__(16) __bf16 Ws[2][128 * 40];   // 40 KB total, 320 KB/WGP avail

    const int tid    = threadIdx.x;
    const int blockM = blockIdx.x * 128;   // M is the fast grid dimension
    const int blockN = blockIdx.y * 128;

    if (blockN + 128 <= N)   // block-uniform: full tile, no N-guards anywhere
        gemm_body<true >(A0, W0, A1, W1, A2, W2, npairs, bias1, bias2, C, N, K, ntC,
                         blockM, blockN, tid, &As[0][0], &As[1][0], &Ws[0][0], &Ws[1][0]);
    else                     // only the single ragged N-column of the logits GEMM
        gemm_body<false>(A0, W0, A1, W1, A2, W2, npairs, bias1, bias2, C, N, K, ntC,
                         blockM, blockN, tid, &As[0][0], &As[1][0], &Ws[0][0], &Ws[1][0]);
}

// ---------------------------------------------------------------------------
extern "C" void kernel_launch(void* const* d_in, const int* in_sizes, int n_in,
                              void* d_out, int out_size, void* d_ws, size_t ws_size,
                              hipStream_t stream) {
    const int*   x   = (const int*)  d_in[0];
    const float* h   = (const float*)d_in[1];
    const float* c   = (const float*)d_in[2];
    const float* emb = (const float*)d_in[3];
    const float* Wxi = (const float*)d_in[4];
    const float* Whi = (const float*)d_in[5];
    const float* bhi = (const float*)d_in[6];
    const float* Wxc = (const float*)d_in[7];
    const float* Whc = (const float*)d_in[8];
    const float* bhc = (const float*)d_in[9];
    const float* Wxo = (const float*)d_in[10];
    const float* Who = (const float*)d_in[11];
    const float* bho = (const float*)d_in[12];
    const float* Wy  = (const float*)d_in[13];
    const float* by  = (const float*)d_in[14];

    float* out    = (float*)d_out;
    float* logits = out;                                  // [B, V]
    float* h_new  = out + (size_t)BB * VV;                // [B, H]
    float* c_new  = h_new + (size_t)BB * HH;              // [B, H]

    float* ws   = (float*)d_ws;
    float* xe   = ws;                                     // [B, E]
    float* pre  = xe   + (size_t)BB * EE;                 // [B, H]
    float* cand = pre  + (size_t)BB * HH;                 // [B, H]
    float* hc   = cand + (size_t)BB * HH;                 // [B, H]
    float* opre = hc   + (size_t)BB * HH;                 // [B, H]

    // 1) xe = emb[x]
    gather_rows<<<BB, 256, 0, stream>>>(x, emb, xe);

    dim3 blk(256);
    dim3 gH(BB / 128, (HH + 127) / 128);   // x = M blocks (fast), y = N blocks

    // 2) pre = xe*Wxi^T + h*Whi^T + c*Whc^T + bhi + bhc
    gemm_bf16<<<gH, blk, 0, stream>>>(xe, Wxi, h, Whi, c, Whc, 3, bhi, bhc, pre, HH, EE, 0);
    // 3) cand = xe*Wxc^T + h*Whc^T + bhc
    gemm_bf16<<<gH, blk, 0, stream>>>(xe, Wxc, h, Whc, nullptr, nullptr, 2, bhc, nullptr,
                                      cand, HH, EE, 0);

    // 4) c_new = tanh(pre)*(c + tanh(cand));  hc = h + c_new
    int n = BB * HH;
    cell_ew<<<(n + 255) / 256, 256, 0, stream>>>(pre, cand, c, h, c_new, hc, n);

    // 5) opre = xe*Wxo^T + (h + c_new)*Who^T + 2*bho
    gemm_bf16<<<gH, blk, 0, stream>>>(xe, Wxo, hc, Who, nullptr, nullptr, 2, bho, bho,
                                      opre, HH, EE, 0);

    // 6) h_new = tanh(opre) * tanh(c_new)
    hnew_ew<<<(n + 255) / 256, 256, 0, stream>>>(opre, c_new, h_new, n);

    // 7) logits = h_new * Wy^T + by  (HBM-bound; NT stores keep Wy/A in L2)
    dim3 gV(BB / 128, (VV + 127) / 128);   // x = M blocks (fast), y = N blocks
    gemm_bf16<<<gV, blk, 0, stream>>>(h_new, Wy, nullptr, nullptr, nullptr, nullptr,
                                      1, by, nullptr, logits, VV, HH, 1);
}
